// SeqAttention_10376640987442
// MI455X (gfx1250) — matrix operand order repaired and checked by reading, not compiled, not run on hardware
//
#include <hip/hip_runtime.h>

// Problem dims (from reference): K batches*heads, M queries, L span, H head dim
#define KB 64
#define MQ 512
#define LW 1024
#define HD 64
#define KVN (MQ + LW)            // 1536 key/value rows

#define BM 64                    // query rows per workgroup (4 waves x 16)
#define WROWS 16                 // query rows per wave
#define BN 32                    // keys per main-loop iteration
#define NIT ((LW + BM) / BN)     // 34 tiles cover band union [m0, m0+L+BM)
#define NTHREADS 128             // 4 wave32s

typedef __attribute__((ext_vector_type(16))) __bf16 v16bf;
typedef __attribute__((ext_vector_type(8)))  __bf16 v8bf;
typedef __attribute__((ext_vector_type(8)))  float  v8f;

// ---- LDS layout (bytes) ----
#define SM_Q     0                         // [64][64] bf16   =   8192
#define SM_POS   (SM_Q   + BM*HD*2)        // [64][1024] bf16 = 131072
#define SM_KB    (SM_POS + BM*LW*2)        // 2 x [32 keys][64 h] bf16 = 8192
#define SM_VTB   (SM_KB  + 2*BN*HD*2)      // 2 x [64 h][32 keys] bf16 = 8192
#define SM_P     (SM_VTB + 2*HD*BN*2)      // 4 waves x [16][32] bf16  = 4096
#define SM_TOTAL (SM_P   + 4*WROWS*BN*2)   // = 159744 B (2 WGs fit 320KB WGP)

// ---- CDNA5 async global->LDS copy (ASYNCcnt) ----
// LDS address of a generic pointer to __shared__ = low 32 bits (flat aperture rule)
__device__ inline void async_copy16(const __bf16* gsrc, __bf16* ldst) {
  asm volatile("global_load_async_to_lds_b128 %0, %1, off"
               :: "v"((unsigned int)(uintptr_t)ldst),
                  "v"((unsigned long long)(uintptr_t)gsrc)
               : "memory");
}
__device__ inline void wait_async0() {
  asm volatile("s_wait_asynccnt 0" ::: "memory");
}

// D = A*B + C, bf16 inputs, f32 accumulate (v_wmma_f32_16x16x32_bf16)
__device__ inline v8f wmma_bf16(v16bf a, v16bf b, v8f c) {
  return __builtin_amdgcn_wmma_f32_16x16x32_bf16(false, a, false, b,
                                                 (short)0, c, false, false);
}

// A fragment (16x32 bf16) from row-major LDS tile.
// A layout: lane holds row m=lane%16; element j -> k = 16*khi + j
__device__ inline v16bf frag_a(const __bf16* s, int stride, int row, int col0, int khi) {
  const __bf16* p = s + row * stride + col0 + 16 * khi;
  v8bf lo = *(const v8bf*)p;
  v8bf hi = *(const v8bf*)(p + 8);
  v16bf r;
#pragma unroll
  for (int j = 0; j < 8; ++j) { r[j] = lo[j]; r[j + 8] = hi[j]; }
  return r;
}

// B fragment (32x16 bf16) where B[k][n] = s[(nrow0 + n)*stride + k0 + k]
// B layout: lane holds column n=lane%16; element j -> k = 16*khi + j
__device__ inline v16bf frag_b(const __bf16* s, int stride, int nrow0, int k0,
                               int khi, int l16) {
  const __bf16* p = s + (nrow0 + l16) * stride + k0 + 16 * khi;
  v8bf lo = *(const v8bf*)p;
  v8bf hi = *(const v8bf*)(p + 8);
  v16bf r;
#pragma unroll
  for (int j = 0; j < 8; ++j) { r[j] = lo[j]; r[j + 8] = hi[j]; }
  return r;
}

// ---- prep: K f32 -> bf16 (same layout) ----
__global__ __launch_bounds__(256)
void cvt_k_kernel(const float* __restrict__ in, __bf16* __restrict__ out) {
  size_t i = ((size_t)blockIdx.x * 256 + threadIdx.x) * 4;
  float4 f = *(const float4*)(in + i);
  out[i]     = (__bf16)f.x;
  out[i + 1] = (__bf16)f.y;
  out[i + 2] = (__bf16)f.z;
  out[i + 3] = (__bf16)f.w;
}

// ---- prep: V f32 [b][n][h] -> bf16 transposed Vt [b][h][n] ----
__global__ __launch_bounds__(256)
void transpose_v_kernel(const float* __restrict__ vin, __bf16* __restrict__ vt) {
  __shared__ __bf16 t[32][72];          // padded vs bank conflicts
  const int b = blockIdx.y, n0 = blockIdx.x * 32;
  const float* src = vin + ((size_t)b * KVN + n0) * HD;
  for (int e = threadIdx.x; e < 32 * 64; e += 256) {
    int n = e >> 6, h = e & 63;
    t[n][h] = (__bf16)src[e];           // coalesced read
  }
  __syncthreads();
  __bf16* dst = vt + (size_t)b * HD * KVN + n0;
  for (int e = threadIdx.x; e < 64 * 32; e += 256) {
    int h = e >> 5, n = e & 31;
    dst[(size_t)h * KVN + n] = t[n][h]; // 64B-contiguous writes per row
  }
}

__global__ __launch_bounds__(NTHREADS)
void seq_attn_kernel(const float* __restrict__ qg, const __bf16* __restrict__ kbg,
                     const __bf16* __restrict__ vtg, const float* __restrict__ peg,
                     float* __restrict__ outg) {
  extern __shared__ char smem[];
  __bf16* sQ    = (__bf16*)(smem + SM_Q);
  __bf16* sPos  = (__bf16*)(smem + SM_POS);
  __bf16* sKbuf = (__bf16*)(smem + SM_KB);
  __bf16* sVbuf = (__bf16*)(smem + SM_VTB);
  __bf16* sP    = (__bf16*)(smem + SM_P);

  const int tid  = threadIdx.x;
  const int wv   = tid >> 5;
  const int lane = tid & 31;
  const int l16  = lane & 15;
  const int khi  = lane >> 4;
  const int b    = blockIdx.x >> 3;          // MQ/BM == 8 blocks per batch
  const int m0   = (blockIdx.x & 7) * BM;

  const float scale = 0.125f;                // 1/sqrt(64), folded into Q
  const float* Qg  = qg  + ((size_t)b * MQ + m0) * HD;
  const __bf16* Kb = kbg + ((size_t)b * KVN + m0) * HD;     // bf16 [n][h]
  const __bf16* Vtb = vtg + (size_t)b * HD * KVN + m0;      // bf16 [h][n]

  // ---- stage Q block (pre-scaled) as bf16 row-major in LDS ----
  for (int e = tid * 2; e < BM * HD; e += NTHREADS * 2) {
    float2 f = *(const float2*)(Qg + e);
    sQ[e]     = (__bf16)(f.x * scale);
    sQ[e + 1] = (__bf16)(f.y * scale);
  }
  __syncthreads();

  // A fragments for ALL four 16-row groups (used for pos; own group for main loop)
  v16bf aq[8];
#pragma unroll
  for (int g = 0; g < 4; ++g) {
    aq[2 * g]     = frag_a(sQ + g * WROWS * HD, HD, l16, 0,  khi);
    aq[2 * g + 1] = frag_a(sQ + g * WROWS * HD, HD, l16, 32, khi);
  }

  // async tile stager: K (row-major) and Vt tiles, 4x b128 per thread
  auto stage = [&](int itx, int buf) {
    const int n0s = itx * BN;
    __bf16* sk = sKbuf + buf * (BN * HD);
    __bf16* sv = sVbuf + buf * (HD * BN);
#pragma unroll
    for (int c = tid; c < 256; c += NTHREADS) {       // K: 32 rows x 8 chunks
      int row = c >> 3, col = (c & 7) << 3;
      async_copy16(Kb + (size_t)(n0s + row) * HD + col, sk + row * HD + col);
    }
#pragma unroll
    for (int c = tid; c < 256; c += NTHREADS) {       // Vt: 64 rows x 4 chunks
      int h = c >> 2, col = (c & 3) << 3;
      async_copy16(Vtb + (size_t)h * KVN + n0s + col, sv + h * BN + col);
    }
  };
  stage(0, 0);  // overlap first tile's DMA with the pos precompute below

  // ---- positional scores pos[m][l] = Qs[m] . pe[:,l] ----
  // wave wv gathers pe B-fragments for its 16 l-tiles ONCE, reused by all 4 row groups
  for (int lt = wv * 16; lt < wv * 16 + 16; ++lt) {
    const int l0 = lt * 16;
    const float* p0 = peg + (size_t)(16 * khi) * LW + l0 + l16;
    const float* p1 = p0 + (size_t)32 * LW;
    v16bf b0, b1;
#pragma unroll
    for (int j = 0; j < 16; ++j) {
      b0[j] = (__bf16)p0[(size_t)j * LW];
      b1[j] = (__bf16)p1[(size_t)j * LW];
    }
#pragma unroll
    for (int g = 0; g < 4; ++g) {
      v8f c = {};
      c = wmma_bf16(aq[2 * g], b0, c);
      c = wmma_bf16(aq[2 * g + 1], b1, c);
#pragma unroll
      for (int r = 0; r < 8; ++r)
        sPos[(g * WROWS + r + 8 * khi) * LW + l0 + l16] = (__bf16)c[r];
    }
  }
  __syncthreads();   // pos rows written cross-wave

  // ---- flash-attention state ----
  v8f o0 = {}, o1 = {}, o2 = {}, o3 = {};
  float mrow[8], lrow[8];
#pragma unroll
  for (int r = 0; r < 8; ++r) { mrow[r] = -1e30f; lrow[r] = 0.f; }

  const int mw0 = wv * WROWS;

  for (int it = 0; it < NIT; ++it) {
    const int cur = it & 1;
    wait_async0();                 // my stage(it) ops landed
    __syncthreads();               // everyone's stage(it) landed; compute(it-1) done
    if (it + 1 < NIT) stage(it + 1, cur ^ 1);   // DMA overlaps compute below

    const int dlt = it * BN - mw0; // l = n - m offset; uniform per wave
    if (dlt + (BN - 1) < 0 || dlt - (WROWS - 1) >= LW) continue; // out of band

    const __bf16* sK  = sKbuf + cur * (BN * HD);
    const __bf16* sVt = sVbuf + cur * (HD * BN);

    // ---- S = Qs @ K^T : two 16x16 n-tiles, k over H=64 ----
    v8f s0 = {}, s1 = {};
    {
      v16bf bk;
      bk = frag_b(sK, HD, 0,  0,  khi, l16); s0 = wmma_bf16(aq[2 * wv],     bk, s0);
      bk = frag_b(sK, HD, 0,  32, khi, l16); s0 = wmma_bf16(aq[2 * wv + 1], bk, s0);
      bk = frag_b(sK, HD, 16, 0,  khi, l16); s1 = wmma_bf16(aq[2 * wv],     bk, s1);
      bk = frag_b(sK, HD, 16, 32, khi, l16); s1 = wmma_bf16(aq[2 * wv + 1], bk, s1);
    }

    // ---- add skewed positional term + band mask ----
#pragma unroll
    for (int r = 0; r < 8; ++r) {
      const int mrel = r + 8 * khi;            // C layout: M = r + 8*(lane/16)
      const int la = dlt + l16 - mrel;
      const int lb = la + 16;
      const __bf16* posrow = sPos + (mw0 + mrel) * LW;
      const bool inA = (la >= 0) & (la < LW);
      const bool inB = (lb >= 0) & (lb < LW);
      float pa = (float)posrow[inA ? la : 0];
      float pb = (float)posrow[inB ? lb : 0];
      s0[r] = inA ? (s0[r] + pa) : -1e30f;
      s1[r] = inB ? (s1[r] + pb) : -1e30f;
    }

    // ---- online softmax (row reductions across 16-lane groups) ----
#pragma unroll
    for (int r = 0; r < 8; ++r) {
      float mx = fmaxf(s0[r], s1[r]);
      mx = fmaxf(mx, __shfl_xor(mx, 1, 16));
      mx = fmaxf(mx, __shfl_xor(mx, 2, 16));
      mx = fmaxf(mx, __shfl_xor(mx, 4, 16));
      mx = fmaxf(mx, __shfl_xor(mx, 8, 16));
      const float mnew  = fmaxf(mrow[r], mx);
      const float alpha = __expf(mrow[r] - mnew);
      const float e0 = __expf(s0[r] - mnew);
      const float e1 = __expf(s1[r] - mnew);
      s0[r] = e0; s1[r] = e1;
      float rs = e0 + e1;
      rs += __shfl_xor(rs, 1, 16);
      rs += __shfl_xor(rs, 2, 16);
      rs += __shfl_xor(rs, 4, 16);
      rs += __shfl_xor(rs, 8, 16);
      lrow[r] = lrow[r] * alpha + rs;
      mrow[r] = mnew;
      o0[r] *= alpha; o1[r] *= alpha; o2[r] *= alpha; o3[r] *= alpha;
    }

    // ---- P (C layout) -> LDS scratch -> A layout, then O += P @ V ----
    __bf16* pw = sP + wv * WROWS * BN;
#pragma unroll
    for (int r = 0; r < 8; ++r) {
      const int mrel = r + 8 * khi;
      pw[mrel * BN + l16]      = (__bf16)s0[r];
      pw[mrel * BN + 16 + l16] = (__bf16)s1[r];
    }
    v16bf ap = frag_a(pw, BN, l16, 0, khi);
    v16bf bv;
    bv = frag_b(sVt, BN, 0,  0, khi, l16); o0 = wmma_bf16(ap, bv, o0);
    bv = frag_b(sVt, BN, 16, 0, khi, l16); o1 = wmma_bf16(ap, bv, o1);
    bv = frag_b(sVt, BN, 32, 0, khi, l16); o2 = wmma_bf16(ap, bv, o2);
    bv = frag_b(sVt, BN, 48, 0, khi, l16); o3 = wmma_bf16(ap, bv, o3);
  }

  // ---- epilogue: normalize and store f32 output ----
  float* Og = outg + ((size_t)b * MQ + m0 + mw0) * HD;
#pragma unroll
  for (int r = 0; r < 8; ++r) {
    const int mrel = r + 8 * khi;
    const float inv = 1.0f / lrow[r];
    float* orow = Og + (size_t)mrel * HD;
    orow[0 * 16 + l16] = o0[r] * inv;
    orow[1 * 16 + l16] = o1[r] * inv;
    orow[2 * 16 + l16] = o2[r] * inv;
    orow[3 * 16 + l16] = o3[r] * inv;
  }
}

extern "C" void kernel_launch(void* const* d_in, const int* in_sizes, int n_in,
                              void* d_out, int out_size, void* d_ws, size_t ws_size,
                              hipStream_t stream) {
  (void)in_sizes; (void)n_in; (void)out_size; (void)ws_size;
  const float* q  = (const float*)d_in[0];
  const float* k  = (const float*)d_in[1];
  const float* v  = (const float*)d_in[2];
  const float* pe = (const float*)d_in[3];
  float* out = (float*)d_out;

  // workspace: bf16 K [b][n][h] then bf16 Vt [b][h][n]  (2 x 12.6MB = 25.2MB)
  const size_t kv_elems = (size_t)KB * KVN * HD;
  __bf16* kbf  = (__bf16*)d_ws;
  __bf16* vtbf = kbf + kv_elems;

  cvt_k_kernel<<<dim3((unsigned)(kv_elems / (256 * 4))), 256, 0, stream>>>(k, kbf);
  transpose_v_kernel<<<dim3(KVN / 32, KB), 256, 0, stream>>>(v, vtbf);

  (void)hipFuncSetAttribute((const void*)seq_attn_kernel,
                            hipFuncAttributeMaxDynamicSharedMemorySize, SM_TOTAL);
  dim3 grid(KB * (MQ / BM));   // 64 batches x 8 query blocks = 512 WGs
  seq_attn_kernel<<<grid, NTHREADS, SM_TOTAL, stream>>>(q, kbf, vtbf, pe, out);
}
